// SwinIRBlock_41343355191388
// MI455X (gfx1250) — matrix operand order repaired and checked by reading, not compiled, not run on hardware
//
#include <hip/hip_runtime.h>
#include <hip/hip_bf16.h>
#include <math.h>

// ---- problem constants ----
#define BB     8
#define CINC   3
#define HH     256
#define WWD    256
#define ED     64
#define NHD    4
#define WS     8
#define SHF    4
#define HDIM   16
#define MLPH   256
#define NTOK   64          // WS*WS
#define NWIN   1024        // (H/WS)*(W/WS)

typedef __attribute__((ext_vector_type(16))) _Float16 v16h;
typedef __attribute__((ext_vector_type(8)))  float    v8f;

// ---------------- WMMA fragment loaders (CDNA5 ISA 7.12.2 layouts) ----------------
// A-matrix 16x32 f16, row-major A[m*lda + k]:
//   lane L: m = m0 + (L&15); kb = k0 + (L>>4)*8
//   halves 0..7  -> K = kb..kb+7 ; halves 8..15 -> K = kb+16..kb+23
__device__ inline v16h frag_a_f16(const _Float16* A, int lda, int m0, int k0, int lane) {
  int m  = m0 + (lane & 15);
  int kb = k0 + ((lane >> 4) << 3);
  const _Float16* p = A + m * lda + kb;
  v16h a;
#pragma unroll
  for (int i = 0; i < 8; ++i) a[i] = p[i];
#pragma unroll
  for (int i = 0; i < 8; ++i) a[8 + i] = p[16 + i];
  return a;
}

__device__ inline v16h frag_a_f32(const float* A, int lda, int m0, int k0, int lane) {
  int m  = m0 + (lane & 15);
  int kb = k0 + ((lane >> 4) << 3);
  const float* p = A + m * lda + kb;
  v16h a;
#pragma unroll
  for (int i = 0; i < 8; ++i) a[i] = (_Float16)p[i];
#pragma unroll
  for (int i = 0; i < 8; ++i) a[8 + i] = (_Float16)p[16 + i];
  return a;
}

// B-matrix 32x16 f16 supplied as B^T row-major BT[n*ldb + k]:
//   lane L: n = n0 + (L&15); kb = k0 + (L>>4)*8; halves 0..7 -> kb..kb+7, 8..15 -> kb+16..kb+23
__device__ inline v16h frag_bt_f16(const _Float16* BT, int ldb, int k0, int n0, int lane) {
  int n  = n0 + (lane & 15);
  int kb = k0 + ((lane >> 4) << 3);
  const _Float16* p = BT + n * ldb + kb;
  v16h b;
#pragma unroll
  for (int i = 0; i < 8; ++i) b[i] = p[i];
#pragma unroll
  for (int i = 0; i < 8; ++i) b[8 + i] = p[16 + i];
  return b;
}

#define WMMA_F16(a, b, c) \
  __builtin_amdgcn_wmma_f32_16x16x32_f16(false, (a), false, (b), (short)0, (c), false, false)

// ---------------- kernel 0: weight prep ----------------
// half buffer: [0) qkv 12288 | [12288) proj 4096 | [16384) fc1 16384 | [32768) fc2 16384
// then fp32:   wu_t[co][kh][kw][ci] (1728) | wet[ci][kh][kw][co] (1728)
__global__ void wconv_kernel(const float* __restrict__ qkv_w, const float* __restrict__ proj_w,
                             const float* __restrict__ fc1_w, const float* __restrict__ fc2_w,
                             const float* __restrict__ conv_u_w, const float* __restrict__ conv_e_w,
                             _Float16* __restrict__ wh, float* __restrict__ wu_t,
                             float* __restrict__ wet) {
  int i = blockIdx.x * 256 + threadIdx.x;
  if (i < 12288)       wh[i] = (_Float16)qkv_w[i];
  else if (i < 16384)  wh[i] = (_Float16)proj_w[i - 12288];
  else if (i < 32768)  wh[i] = (_Float16)fc1_w[i - 16384];
  else if (i < 49152)  wh[i] = (_Float16)fc2_w[i - 32768];
  else if (i < 50880) {                       // conv_u (3,64,3,3) -> [co][kh][kw][ci]
    int j  = i - 49152;
    int ci = j & 63;
    int r  = j >> 6;                          // co*9 + kh*3 + kw
    int kw = r % 3, kh = (r / 3) % 3, co = r / 9;
    wu_t[j] = conv_u_w[((co * ED + ci) * 3 + kh) * 3 + kw];
  } else if (i < 52608) {                     // conv_e (64,3,3,3) -> [ci][kh][kw][co]
    int j  = i - 50880;
    int co = j & 63;
    int r  = j >> 6;                          // ci*9 + kh*3 + kw
    int kw = r % 3, kh = (r / 3) % 3, ci = r / 9;
    wet[j] = conv_e_w[((co * CINC + ci) * 3 + kh) * 3 + kw];
  }
}

// ---------------- kernel 1: conv3x3 embed (NCHW 3ch -> NHWC 64ch) ----------------
__global__ __launch_bounds__(256) void conv_embed_kernel(const float* __restrict__ x,
                                                         const float* __restrict__ wet,
                                                         const float* __restrict__ be,
                                                         float* __restrict__ xe) {
  size_t idx = (size_t)blockIdx.x * 256 + threadIdx.x;   // B*H*W*ED
  int co = (int)(idx & 63);
  size_t p = idx >> 6;
  int w = (int)(p & 255); p >>= 8;
  int h = (int)(p & 255);
  int b = (int)(p >> 8);
  float acc = be[co];
#pragma unroll
  for (int ci = 0; ci < CINC; ++ci)
#pragma unroll
    for (int kh = 0; kh < 3; ++kh) {
      int hy = h + kh - 1;
      if (hy < 0 || hy >= HH) continue;
#pragma unroll
      for (int kw = 0; kw < 3; ++kw) {
        int wx = w + kw - 1;
        if (wx < 0 || wx >= WWD) continue;
        acc += x[(((size_t)b * CINC + ci) * HH + hy) * WWD + wx] *
               wet[(ci * 9 + kh * 3 + kw) * 64 + co];     // coalesced over co
      }
    }
  xe[idx] = acc;
}

// ---------------- kernel 2: fused shifted-window attention (in-place on xe) ----------------
__global__ __launch_bounds__(256) void attn_kernel(float* __restrict__ xe,
                                                   const float* __restrict__ n1_g,
                                                   const float* __restrict__ n1_b,
                                                   const _Float16* __restrict__ qkvw,
                                                   const float* __restrict__ qkv_b,
                                                   const _Float16* __restrict__ projw,
                                                   const float* __restrict__ proj_b) {
  const int tid  = threadIdx.x;
  const int lane = tid & 31;
  // wave index is uniform across the wave: pin it to an SGPR so tile-routing
  // branches become scalar (s_cbranch) and EXEC stays all-ones around WMMA.
  const int wave = __builtin_amdgcn_readfirstlane(tid >> 5);
  const int widx = blockIdx.x;
  const int b  = widx >> 10;
  const int wh = (widx >> 5) & 31;
  const int ww = widx & 31;

  // 64 KiB LDS, manually partitioned + reused
  __shared__ __align__(16) char smem[65536];
  _Float16* Qh    = (_Float16*)(smem);            // [4][64][32] (K padded to 32, upper 16 zero)
  _Float16* Kh    = (_Float16*)(smem + 16384);    // [4][64][32]
  _Float16* Vt    = (_Float16*)(smem + 32768);    // [4][16][64]  (V transposed per head)
  _Float16* xn    = (_Float16*)(smem + 40960);    // [64][64]  -> reused as outh later
  float*    attnf = (float*)   (smem + 49152);    // [64][64] fp32 logits/probs (per head)

  // prefetch weight matrices (gfx1250 global_prefetch_b8), 128B per lane covers both
  __builtin_prefetch((const char*)qkvw + tid * 128, 0, 3);   // 24 KiB
  if (tid < 64) __builtin_prefetch((const char*)projw + tid * 128, 0, 3);  // 8 KiB

  // zero Q/K (covers the zero-padded K range [16,32))
  {
    uint32_t* z = (uint32_t*)smem;
    for (int i = tid; i < 8192; i += 256) z[i] = 0u;
  }

  // ---- phase A: gather window tokens (with -SHIFT roll) + LayerNorm -> xn f16 ----
  if (tid < NTOK) {
    int r = tid >> 3, c = tid & 7;
    int hp = (wh * WS + r + SHF) & (HH - 1);
    int wp = (ww * WS + c + SHF) & (WWD - 1);
    const float4* px4 = (const float4*)(xe + (((size_t)b * HH + hp) * WWD + wp) * ED);
    float4 v4[16];
    float m = 0.f;
#pragma unroll
    for (int k = 0; k < 16; ++k) {
      v4[k] = px4[k];
      m += v4[k].x + v4[k].y + v4[k].z + v4[k].w;
    }
    m *= (1.f / ED);
    float v = 0.f;
#pragma unroll
    for (int k = 0; k < 16; ++k) {
      float dx = v4[k].x - m, dy = v4[k].y - m, dz = v4[k].z - m, dw = v4[k].w - m;
      v += dx * dx + dy * dy + dz * dz + dw * dw;
    }
    v *= (1.f / ED);
    float rstd = rsqrtf(v + 1e-5f);
#pragma unroll
    for (int k = 0; k < 16; ++k) {
      xn[tid * ED + 4 * k + 0] = (_Float16)((v4[k].x - m) * rstd * n1_g[4 * k + 0] + n1_b[4 * k + 0]);
      xn[tid * ED + 4 * k + 1] = (_Float16)((v4[k].y - m) * rstd * n1_g[4 * k + 1] + n1_b[4 * k + 1]);
      xn[tid * ED + 4 * k + 2] = (_Float16)((v4[k].z - m) * rstd * n1_g[4 * k + 2] + n1_b[4 * k + 2]);
      xn[tid * ED + 4 * k + 3] = (_Float16)((v4[k].w - m) * rstd * n1_g[4 * k + 3] + n1_b[4 * k + 3]);
    }
  }
  __syncthreads();

  // ---- phase B: QKV = xn(64x64) @ Wqkv^T(64x192) + b ; scatter into Qh/Kh/Vt ----
  // routing depends only on the (scalar) output tile index `to`:
  //   to<4 -> Q head=to ; 4<=to<8 -> K head=to-4 ; to>=8 -> V head=to-8
  for (int t = wave; t < 48; t += 8) {       // 4 (token tiles) x 12 (out tiles)
    int tm = t & 3, to = t >> 2;
    v8f acc = {};
    acc = WMMA_F16(frag_a_f16(xn, ED, tm * 16, 0,  lane), frag_bt_f16(qkvw, ED, 0,  to * 16, lane), acc);
    acc = WMMA_F16(frag_a_f16(xn, ED, tm * 16, 32, lane), frag_bt_f16(qkvw, ED, 32, to * 16, lane), acc);
    int col = lane & 15;
    float bias = qkv_b[to * 16 + col];
    int mb = tm * 16 + ((lane >> 4) << 3);
    if (to < 4) {                 // Q (scaled by HD^-0.5 = 0.25)
      _Float16* q = Qh + to * 64 * 32;
#pragma unroll
      for (int j = 0; j < 8; ++j) q[(mb + j) * 32 + col] = (_Float16)((acc[j] + bias) * 0.25f);
    } else if (to < 8) {          // K
      _Float16* kk = Kh + (to - 4) * 64 * 32;
#pragma unroll
      for (int j = 0; j < 8; ++j) kk[(mb + j) * 32 + col] = (_Float16)(acc[j] + bias);
    } else {                      // V, transposed
      _Float16* vv = Vt + (to - 8) * 16 * 64 + col * 64;
#pragma unroll
      for (int j = 0; j < 8; ++j) vv[mb + j] = (_Float16)(acc[j] + bias);
    }
  }
  __syncthreads();

  _Float16* outh = xn;   // xn dead; reuse as attention output [64][64] f16

  // ---- phase C: per head QK^T -> masked softmax -> attn @ V ----
  for (int hd = 0; hd < NHD; ++hd) {
    // QK^T : 16 tiles, 2 per wave (K padded to 32 with zeros -> single WMMA)
    for (int t = wave * 2; t < wave * 2 + 2; ++t) {
      int tn = t >> 2, tm = t & 3;
      v8f acc = {};
      acc = WMMA_F16(frag_a_f16(Qh + hd * 64 * 32, 32, tn * 16, 0, lane),
                     frag_bt_f16(Kh + hd * 64 * 32, 32, 0, tm * 16, lane), acc);
      int ncol = tm * 16 + (lane & 15);
#pragma unroll
      for (int j = 0; j < 8; ++j) {
        int m = tn * 16 + ((lane >> 4) << 3) + j;
        attnf[m * 64 + ncol] = acc[j];
      }
    }
    __syncthreads();

    // masked softmax, one thread per row, in place on attnf (row-private)
    if (tid < NTOK) {
      int r = tid >> 3, c = tid & 7;
      int hp = wh * WS + r, wp = ww * WS + c;
      int gh = (hp < HH - WS) ? 0 : ((hp < HH - SHF) ? 1 : 2);
      int gw = (wp < WWD - WS) ? 0 : ((wp < WWD - SHF) ? 1 : 2);
      int gn = gh * 3 + gw;
      float* row = attnf + tid * 64;
      float mx = -1e30f;
      for (int m = 0; m < 64; ++m) {
        int hpm = wh * WS + (m >> 3), wpm = ww * WS + (m & 7);
        int ghm = (hpm < HH - WS) ? 0 : ((hpm < HH - SHF) ? 1 : 2);
        int gwm = (wpm < WWD - WS) ? 0 : ((wpm < WWD - SHF) ? 1 : 2);
        float l = row[m] + (((ghm * 3 + gwm) != gn) ? -100.f : 0.f);
        row[m] = l;
        mx = fmaxf(mx, l);
      }
      float s = 0.f;
      for (int m = 0; m < 64; ++m) { float e = __expf(row[m] - mx); row[m] = e; s += e; }
      float inv = 1.f / s;
      for (int m = 0; m < 64; ++m) row[m] *= inv;
    }
    __syncthreads();

    // out_head(64x16) = attn(64x64) @ V(64x16) ; waves 0..3, one token-tile each
    if (wave < 4) {
      int tn = wave;
      v8f acc = {};
      acc = WMMA_F16(frag_a_f32(attnf, 64, tn * 16, 0,  lane), frag_bt_f16(Vt + hd * 16 * 64, 64, 0,  0, lane), acc);
      acc = WMMA_F16(frag_a_f32(attnf, 64, tn * 16, 32, lane), frag_bt_f16(Vt + hd * 16 * 64, 64, 32, 0, lane), acc);
      int d = lane & 15;
#pragma unroll
      for (int j = 0; j < 8; ++j) {
        int m = tn * 16 + ((lane >> 4) << 3) + j;
        outh[m * 64 + hd * 16 + d] = (_Float16)acc[j];
      }
    }
    __syncthreads();   // protects attnf reuse next head + outh before proj
  }

  // ---- phase D: proj + residual, written in place to xe ----
  for (int t = wave * 2; t < wave * 2 + 2; ++t) {
    int tm = t >> 2, to = t & 3;
    v8f acc = {};
    acc = WMMA_F16(frag_a_f16(outh, ED, tm * 16, 0,  lane), frag_bt_f16(projw, ED, 0,  to * 16, lane), acc);
    acc = WMMA_F16(frag_a_f16(outh, ED, tm * 16, 32, lane), frag_bt_f16(projw, ED, 32, to * 16, lane), acc);
    int o = to * 16 + (lane & 15);
    float pb = proj_b[o];
#pragma unroll
    for (int j = 0; j < 8; ++j) {
      int m = tm * 16 + ((lane >> 4) << 3) + j;
      int hp = (wh * WS + (m >> 3) + SHF) & (HH - 1);
      int wp = (ww * WS + (m & 7) + SHF) & (WWD - 1);
      float* px = xe + (((size_t)b * HH + hp) * WWD + wp) * ED + o;
      *px = *px + acc[j] + pb;    // (m,o) owned by exactly this lane/j -> race-free
    }
  }
}

// ---------------- kernel 3: fused MLP (LN -> fc1 -> GeLU -> fc2) + residual, in place ----------------
__global__ __launch_bounds__(256) void mlp_kernel(float* __restrict__ xio,
                                                  const float* __restrict__ n2_g,
                                                  const float* __restrict__ n2_b,
                                                  const _Float16* __restrict__ fc1w,
                                                  const float* __restrict__ fc1_b,
                                                  const _Float16* __restrict__ fc2w,
                                                  const float* __restrict__ fc2_b) {
  const int tid  = threadIdx.x;
  const int lane = tid & 31;
  const int wave = __builtin_amdgcn_readfirstlane(tid >> 5);
  const size_t t0 = (size_t)blockIdx.x * 64;   // 64 tokens per block

  __shared__ __align__(16) char smem[40960];
  _Float16* xnm = (_Float16*)(smem);           // [64][64]
  _Float16* hm  = (_Float16*)(smem + 8192);    // [64][256]

  __builtin_prefetch((const char*)fc1w + tid * 128, 0, 3);   // 32 KiB
  __builtin_prefetch((const char*)fc2w + tid * 128, 0, 3);   // 32 KiB

  if (tid < 64) {
    const float4* px4 = (const float4*)(xio + (t0 + tid) * ED);
    float4 v4[16];
    float m = 0.f;
#pragma unroll
    for (int k = 0; k < 16; ++k) {
      v4[k] = px4[k];
      m += v4[k].x + v4[k].y + v4[k].z + v4[k].w;
    }
    m *= (1.f / ED);
    float v = 0.f;
#pragma unroll
    for (int k = 0; k < 16; ++k) {
      float dx = v4[k].x - m, dy = v4[k].y - m, dz = v4[k].z - m, dw = v4[k].w - m;
      v += dx * dx + dy * dy + dz * dz + dw * dw;
    }
    v *= (1.f / ED);
    float rstd = rsqrtf(v + 1e-5f);
#pragma unroll
    for (int k = 0; k < 16; ++k) {
      xnm[tid * ED + 4 * k + 0] = (_Float16)((v4[k].x - m) * rstd * n2_g[4 * k + 0] + n2_b[4 * k + 0]);
      xnm[tid * ED + 4 * k + 1] = (_Float16)((v4[k].y - m) * rstd * n2_g[4 * k + 1] + n2_b[4 * k + 1]);
      xnm[tid * ED + 4 * k + 2] = (_Float16)((v4[k].z - m) * rstd * n2_g[4 * k + 2] + n2_b[4 * k + 2]);
      xnm[tid * ED + 4 * k + 3] = (_Float16)((v4[k].w - m) * rstd * n2_g[4 * k + 3] + n2_b[4 * k + 3]);
    }
  }
  __syncthreads();

  // fc1: (64x64) @ (64x256) -> GeLU -> hm ; 64 tiles
  for (int t = wave; t < 64; t += 8) {
    int tm = t & 3, to = t >> 2;
    v8f acc = {};
    acc = WMMA_F16(frag_a_f16(xnm, ED, tm * 16, 0,  lane), frag_bt_f16(fc1w, ED, 0,  to * 16, lane), acc);
    acc = WMMA_F16(frag_a_f16(xnm, ED, tm * 16, 32, lane), frag_bt_f16(fc1w, ED, 32, to * 16, lane), acc);
    int o = to * 16 + (lane & 15);
    float bb = fc1_b[o];
#pragma unroll
    for (int j = 0; j < 8; ++j) {
      int m = tm * 16 + ((lane >> 4) << 3) + j;
      float v = acc[j] + bb;
      float g = 0.5f * v * (1.f + erff(v * 0.70710678f));   // exact GeLU
      hm[m * MLPH + o] = (_Float16)g;
    }
  }
  __syncthreads();

  // fc2: (64x256) @ (256x64) + residual, in place ; 16 tiles
  for (int t = wave * 2; t < wave * 2 + 2; ++t) {
    int tm = t >> 2, to = t & 3;
    v8f acc = {};
#pragma unroll
    for (int kk = 0; kk < 8; ++kk)
      acc = WMMA_F16(frag_a_f16(hm, MLPH, tm * 16, kk * 32, lane),
                     frag_bt_f16(fc2w, MLPH, kk * 32, to * 16, lane), acc);
    int o = to * 16 + (lane & 15);
    float bb = fc2_b[o];
#pragma unroll
    for (int j = 0; j < 8; ++j) {
      int m = tm * 16 + ((lane >> 4) << 3) + j;
      float* px = xio + (t0 + m) * ED + o;
      *px = *px + acc[j] + bb;
    }
  }
}

// ---------------- kernel 4: conv3x3 out (NHWC 64ch -> NCHW 3ch) + input residual ----------------
__global__ __launch_bounds__(256) void conv_out_kernel(const float* __restrict__ xf,
                                                       const float* __restrict__ wu_t,
                                                       const float* __restrict__ bu,
                                                       const float* __restrict__ xin,
                                                       float* __restrict__ out) {
  size_t idx = (size_t)blockIdx.x * 256 + threadIdx.x;   // B*3*H*W
  int w = (int)(idx & 255); size_t p = idx >> 8;
  int h = (int)(p & 255);   p >>= 8;
  int co = (int)(p % 3);
  int b  = (int)(p / 3);
  float acc = bu[co];
#pragma unroll
  for (int kh = 0; kh < 3; ++kh) {
    int hy = h + kh - 1;
    if (hy < 0 || hy >= HH) continue;
#pragma unroll
    for (int kw = 0; kw < 3; ++kw) {
      int wx = w + kw - 1;
      if (wx < 0 || wx >= WWD) continue;
      const float4* prow = (const float4*)(xf + (((size_t)b * HH + hy) * WWD + wx) * ED);
      const float4* wrow = (const float4*)(wu_t + (co * 9 + kh * 3 + kw) * 64);
      for (int ci = 0; ci < 16; ++ci) {
        float4 a4 = prow[ci], b4 = wrow[ci];
        acc += a4.x * b4.x + a4.y * b4.y + a4.z * b4.z + a4.w * b4.w;
      }
    }
  }
  out[idx] = xin[idx] + acc;
}

// ---------------- launcher ----------------
extern "C" void kernel_launch(void* const* d_in, const int* in_sizes, int n_in,
                              void* d_out, int out_size, void* d_ws, size_t ws_size,
                              hipStream_t stream) {
  const float* x        = (const float*)d_in[0];
  const float* conv_e_w = (const float*)d_in[1];
  const float* conv_e_b = (const float*)d_in[2];
  const float* n1_g     = (const float*)d_in[3];
  const float* n1_b     = (const float*)d_in[4];
  const float* qkv_w    = (const float*)d_in[5];
  const float* qkv_b    = (const float*)d_in[6];
  const float* proj_w   = (const float*)d_in[7];
  const float* proj_b   = (const float*)d_in[8];
  const float* n2_g     = (const float*)d_in[9];
  const float* n2_b     = (const float*)d_in[10];
  const float* fc1_w    = (const float*)d_in[11];
  const float* fc1_b    = (const float*)d_in[12];
  const float* fc2_w    = (const float*)d_in[13];
  const float* fc2_b    = (const float*)d_in[14];
  const float* conv_u_w = (const float*)d_in[15];
  const float* conv_u_b = (const float*)d_in[16];

  const size_t feat_bytes = (size_t)BB * HH * WWD * ED * sizeof(float);   // 128 MiB
  float*    xe   = (float*)d_ws;
  _Float16* wh   = (_Float16*)((char*)d_ws + feat_bytes);                 // 49152 halves
  float*    wu_t = (float*)((char*)d_ws + feat_bytes + 98304);            // 1728 floats
  float*    wet  = wu_t + 1728;                                           // 1728 floats
  const _Float16* qkvw_h  = wh;
  const _Float16* projw_h = wh + 12288;
  const _Float16* fc1w_h  = wh + 16384;
  const _Float16* fc2w_h  = wh + 32768;

  wconv_kernel<<<206, 256, 0, stream>>>(qkv_w, proj_w, fc1_w, fc2_w, conv_u_w, conv_e_w,
                                        wh, wu_t, wet);
  conv_embed_kernel<<<(BB * HH * WWD * ED) / 256, 256, 0, stream>>>(x, wet, conv_e_b, xe);
  attn_kernel<<<BB * NWIN, 256, 0, stream>>>(xe, n1_g, n1_b, qkvw_h, qkv_b, projw_h, proj_b);
  mlp_kernel<<<(BB * HH * WWD) / 64, 256, 0, stream>>>(xe, n2_g, n2_b, fc1w_h, fc1_b, fc2w_h, fc2_b);
  conv_out_kernel<<<(BB * CINC * HH * WWD) / 256, 256, 0, stream>>>(xe, wu_t, conv_u_b, x, (float*)d_out);
}
